// HybridMemory_88321707475308
// MI455X (gfx1250) — matrix operand order repaired and checked by reading, not compiled, not run on hardware
//
#include <hip/hip_runtime.h>
#include <hip/hip_bf16.h>

typedef __bf16 bf16;
typedef __attribute__((ext_vector_type(16))) __bf16 v16bf;
typedef __attribute__((ext_vector_type(8)))  __bf16 v8bf;
typedef __attribute__((ext_vector_type(8)))  float   v8f;
typedef __attribute__((ext_vector_type(4)))  float   v4f;

#define B_SZ 64
#define D_SZ 2048
#define N_SZ 50000
#define L_SZ 1000
#define TEMP_INV 20.0f   // 1/0.05
#define MOM 0.2f
#define EPSV 1e-6f

// ---------------- init: zero accumulators + loss scalar ----------------
__global__ void k_init(float* __restrict__ simAcc, float* __restrict__ memAcc,
                       int* __restrict__ counts, float* __restrict__ out0) {
    int gid = blockIdx.x * blockDim.x + threadIdx.x;
    int total = L_SZ * B_SZ + B_SZ * D_SZ + L_SZ + 1;
    for (int i = gid; i < total; i += gridDim.x * blockDim.x) {
        if (i < L_SZ * B_SZ)                       simAcc[i] = 0.0f;
        else if (i < L_SZ * B_SZ + B_SZ * D_SZ)    memAcc[i - L_SZ * B_SZ] = 0.0f;
        else if (i < L_SZ * B_SZ + B_SZ * D_SZ + L_SZ)
            counts[i - L_SZ * B_SZ - B_SZ * D_SZ] = 0;
        else                                       *out0 = 0.0f;
    }
}

// ---------------- prep: inputs -> bf16, gather targets ----------------
__global__ void k_prep(const float* __restrict__ inputs, const int* __restrict__ indexes,
                       const int* __restrict__ labels, bf16* __restrict__ inb,
                       int* __restrict__ targets) {
    int gid = blockIdx.x * blockDim.x + threadIdx.x;
    if (gid < B_SZ * D_SZ) inb[gid] = (bf16)inputs[gid];
    if (gid < B_SZ) targets[gid] = labels[indexes[gid]];
}

// ---------------- per-cluster member counts ----------------
__global__ void k_counts(const int* __restrict__ labels, int* __restrict__ counts) {
    int gid = blockIdx.x * blockDim.x + threadIdx.x;
    if (gid < N_SZ) atomicAdd(&counts[labels[gid]], 1);
}

// ---------------- WMMA GEMM + fused label scatter ----------------
// grid: ceil(N/64) blocks, 128 threads (4 waves). Waves split the N (feature-row)
// dimension: wave w owns feature rows [64*bx + 16w, +16); each wave computes all
// four 16-row M tiles (M=64) against its single B fragment -> 4 WMMAs / K-step.
// The 64x32 f32 feature tile is double-buffered through LDS with
// global_load_async_to_lds_b128 (ASYNCcnt pipeline).
__global__ void __launch_bounds__(128) k_gemm(const bf16* __restrict__ inb,
                                              const float* __restrict__ features,
                                              const int* __restrict__ labels,
                                              float* __restrict__ simAcc) {
    __shared__ __align__(16) float btile[2][64 * 32];   // 2 x 8KB ping-pong

    const int tid  = threadIdx.x;
    const int lane = tid & 31;
    const int wave = tid >> 5;
    const int half = lane >> 4;      // 0: lanes 0-15, 1: lanes 16-31
    const int l15  = lane & 15;
    const int kA0  = half * 16;      // A K-offset per lane half
    const int kB0  = half * 16;      // B K-offset per lane half
    const int rowBase = blockIdx.x * 64;

    const uint32_t ldsbase = (uint32_t)(uintptr_t)(&btile[0][0]);

    // cooperative async stage of one 64-row x 32-K f32 tile (8KB) into LDS buf
    auto issue_tile = [&](int kk, int buf) {
#pragma unroll
        for (int i = 0; i < 4; ++i) {
            const int c   = tid + 128 * i;          // 512 chunks of 16B
            int grow      = rowBase + (c >> 3);     // 8 chunks per 128B row
            if (grow > N_SZ - 1) grow = N_SZ - 1;   // tail clamp (loads stay valid)
            const int col = (c & 7) * 4;
            const float* gp = features + (size_t)grow * D_SZ + kk + col;
            const uint32_t lo = ldsbase + (uint32_t)(buf * 8192 + c * 16);
            const uint64_t ga = (uint64_t)(uintptr_t)gp;
            asm volatile("global_load_async_to_lds_b128 %0, %1, off"
                         :: "v"(lo), "v"(ga) : "memory");
        }
    };

    v8f acc[4] = {};
    issue_tile(0, 0);
    for (int kk = 0; kk < D_SZ; kk += 32) {
        const int cur = (kk >> 5) & 1;
        const int nkk = (kk + 32 < D_SZ) ? kk + 32 : 0;   // wrap issue is harmless
        issue_tile(nkk, cur ^ 1);
        // in-order async completion: <=4 outstanding means current tile landed
        asm volatile("s_wait_asynccnt 0x4" ::: "memory");
        __syncthreads();

        // B fragment from LDS: 16 consecutive K f32 -> bf16
        const float* bl = &btile[cur][(wave * 16 + l15) * 32 + kB0];
        v16bf b;
#pragma unroll
        for (int t = 0; t < 4; ++t) {
            v4f x = *reinterpret_cast<const v4f*>(bl + 4 * t);
#pragma unroll
            for (int j = 0; j < 4; ++j) b[4 * t + j] = (bf16)x[j];
        }
        // A fragments (bf16, L2-resident) + 4 WMMAs sharing B
#pragma unroll
        for (int mt = 0; mt < 4; ++mt) {
            const bf16* ap = inb + (mt * 16 + l15) * D_SZ + kk + kA0;
            v8bf alo = *reinterpret_cast<const v8bf*>(ap);
            v8bf ahi = *reinterpret_cast<const v8bf*>(ap + 8);
            v16bf a;
#pragma unroll
            for (int i = 0; i < 8; ++i) { a[i] = alo[i]; a[i + 8] = ahi[i]; }
            acc[mt] = __builtin_amdgcn_wmma_f32_16x16x32_bf16(false, a, false, b,
                                                              (short)0, acc[mt],
                                                              false, false);
        }
        __syncthreads();   // all reads of `cur` done before its buffer is re-filled
    }

    // scatter logits tile into per-label accumulator: sim[l, m] += logit/TEMP
    const int gnrow = rowBase + wave * 16 + l15;
    if (gnrow < N_SZ) {
        const int lbl = labels[gnrow];
        float* dst = simAcc + lbl * B_SZ;
#pragma unroll
        for (int mt = 0; mt < 4; ++mt)
#pragma unroll
            for (int r = 0; r < 8; ++r)
                atomicAdd(&dst[mt * 16 + half * 8 + r], acc[mt][r] * TEMP_INV);
    }
}

// ---------------- bank copy fused with target-cluster feature sums ----------------
__global__ void __launch_bounds__(256) k_copy_mem(const float* __restrict__ features,
                                                  const int* __restrict__ labels,
                                                  const int* __restrict__ targets,
                                                  float* __restrict__ memAcc,
                                                  float* __restrict__ bank) {
    __shared__ int s_n;
    __shared__ int s_lbl;
    __shared__ int s_slot[64];
    const int row = blockIdx.x;
    const int tid = threadIdx.x;
    if (tid == 0) { s_lbl = labels[row]; s_n = 0; }
    __syncthreads();
    if (tid < 64 && targets[tid] == s_lbl) { int p = atomicAdd(&s_n, 1); s_slot[p] = tid; }
    __syncthreads();
    const int nm = s_n;
    const float* src = features + (size_t)row * D_SZ;
    float* dst = bank + (size_t)row * D_SZ;
#pragma unroll
    for (int k = 0; k < 8; ++k) {
        int d = tid + 256 * k;
        float v = src[d];
        dst[d] = v;
        for (int j = 0; j < nm; ++j) atomicAdd(&memAcc[s_slot[j] * D_SZ + d], v);
    }
}

// ---------------- focal loss over masked softmax (one block per batch row) ----------------
__global__ void __launch_bounds__(128) k_focal(const float* __restrict__ simAcc,
                                               const int* __restrict__ counts,
                                               const int* __restrict__ targets,
                                               float* __restrict__ out0) {
    const int b = blockIdx.x;
    const int tid = threadIdx.x;
    const int tgt = targets[b];
    __shared__ float s_sum[128];
    __shared__ float s_p;
    if (tid == 0) s_p = 0.0f;
    __syncthreads();
    float local = 0.0f, pn = -1.0f;
    for (int l = tid; l < L_SZ; l += 128) {
        int c = counts[l];
        if (c > 0) {
            float s = simAcc[l * B_SZ + b] / (float)c;
            float e = expf(s);           // reference: no max-subtraction
            local += e;
            if (l == tgt) pn = e;
        }
    }
    s_sum[tid] = local;
    if (pn >= 0.0f) s_p = pn;
    __syncthreads();
    for (int s = 64; s > 0; s >>= 1) {
        if (tid < s) s_sum[tid] += s_sum[tid + s];
        __syncthreads();
    }
    if (tid == 0) {
        float p  = s_p / (s_sum[0] + EPSV);
        float om = 1.0f - p;
        float f4 = om * om; f4 *= f4;
        float loss = -f4 * logf(p + EPSV);
        atomicAdd(out0, loss * (1.0f / B_SZ));
    }
}

// ---------------- cosine losses: l_memo + l_con (one block per batch row) ----------------
__global__ void __launch_bounds__(256) k_pair(const float* __restrict__ inputs,
                                              const float* __restrict__ another,
                                              const float* __restrict__ memAcc,
                                              float* __restrict__ out0) {
    const int b = blockIdx.x;
    const int tid = threadIdx.x;
    __shared__ float r[5][256];
    float dmc = 0, nm = 0, na = 0, dic = 0, ni = 0;
    for (int d = tid; d < D_SZ; d += 256) {
        float m  = memAcc[b * D_SZ + d];   // 1/denom scale cancels under normalization
        float an = another[b * D_SZ + d];
        float x  = inputs[b * D_SZ + d];
        dmc += m * an; nm += m * m; na += an * an; dic += x * an; ni += x * x;
    }
    r[0][tid] = dmc; r[1][tid] = nm; r[2][tid] = na; r[3][tid] = dic; r[4][tid] = ni;
    __syncthreads();
    for (int s = 128; s > 0; s >>= 1) {
        if (tid < s)
#pragma unroll
            for (int q = 0; q < 5; ++q) r[q][tid] += r[q][tid + s];
        __syncthreads();
    }
    if (tid == 0) {
        float l = -(r[0][0] * rsqrtf(r[1][0] * r[2][0]))
                  - (r[3][0] * rsqrtf(r[4][0] * r[2][0]));
        atomicAdd(out0, l * (1.0f / B_SZ));
    }
}

// ---------------- sequential momentum bank update (duplicate-safe) ----------------
__global__ void __launch_bounds__(256) k_update(const float* __restrict__ inputs,
                                                const int* __restrict__ indexes,
                                                float* __restrict__ bank) {
    __shared__ float red[256];
    const int tid = threadIdx.x;
    for (int i = 0; i < B_SZ; ++i) {
        const int y = indexes[i];
        float vals[8];
        float local = 0.0f;
#pragma unroll
        for (int k = 0; k < 8; ++k) {
            int d = tid + 256 * k;
            float v = MOM * bank[(size_t)y * D_SZ + d] + (1.0f - MOM) * inputs[i * D_SZ + d];
            vals[k] = v;
            local += v * v;
        }
        red[tid] = local;
        __syncthreads();
        for (int s = 128; s > 0; s >>= 1) {
            if (tid < s) red[tid] += red[tid + s];
            __syncthreads();
        }
        float inv = rsqrtf(red[0]);
#pragma unroll
        for (int k = 0; k < 8; ++k)
            bank[(size_t)y * D_SZ + tid + 256 * k] = vals[k] * inv;
        __syncthreads();
    }
}

extern "C" void kernel_launch(void* const* d_in, const int* in_sizes, int n_in,
                              void* d_out, int out_size, void* d_ws, size_t ws_size,
                              hipStream_t stream) {
    const float* inputs   = (const float*)d_in[0];
    const float* another  = (const float*)d_in[1];
    const int*   indexes  = (const int*)d_in[2];
    const float* features = (const float*)d_in[3];
    const int*   labels   = (const int*)d_in[4];
    // d_in[5] = back (always 1 per setup)

    float* out = (float*)d_out;           // out[0] = loss, out+1 = new_features
    char* ws = (char*)d_ws;
    float* simAcc  = (float*)ws;                          // [1000*64] f32
    float* memAcc  = simAcc + L_SZ * B_SZ;                // [64*2048] f32
    int*   counts  = (int*)(memAcc + B_SZ * D_SZ);        // [1000] i32
    int*   targets = counts + L_SZ;                       // [64] i32
    bf16*  inb     = (bf16*)(targets + B_SZ);             // [64*2048] bf16 (16B-aligned)

    k_init<<<256, 256, 0, stream>>>(simAcc, memAcc, counts, out);
    k_prep<<<(B_SZ * D_SZ + 255) / 256, 256, 0, stream>>>(inputs, indexes, labels, inb, targets);
    k_counts<<<(N_SZ + 255) / 256, 256, 0, stream>>>(labels, counts);
    k_gemm<<<(N_SZ + 63) / 64, 128, 0, stream>>>(inb, features, labels, simAcc);
    k_copy_mem<<<N_SZ, 256, 0, stream>>>(features, labels, targets, memAcc, out + 1);
    k_focal<<<B_SZ, 128, 0, stream>>>(simAcc, counts, targets, out);
    k_pair<<<B_SZ, 256, 0, stream>>>(inputs, another, memAcc, out);
    k_update<<<1, 256, 0, stream>>>(inputs, indexes, out + 1);
}
